// SelfAttention_30391188587267
// MI455X (gfx1250) — matrix operand order repaired and checked by reading, not compiled, not run on hardware
//
#include <hip/hip_runtime.h>
#include <hip/hip_bf16.h>

typedef _Float16 half_t;
typedef __attribute__((ext_vector_type(16))) _Float16 v16h;
typedef __attribute__((ext_vector_type(4)))  _Float16 v4h;
typedef __attribute__((ext_vector_type(8)))  float    v8f;

#define HIDDEN   1024
#define HEADS    16
#define HEAD_DIM 64
#define SEQ      2048
#define BATCH    4
#define MTOT     (BATCH * SEQ)

// ---------------------------------------------------------------------------
// CDNA5 async global->LDS helpers (ASYNCcnt-tracked, no VGPR data round-trip)
// ---------------------------------------------------------------------------
__device__ __forceinline__ unsigned lds_addr_of(const void* p) {
  // generic LDS pointer: low 32 bits are the LDS byte offset (ISA §10.2)
  return (unsigned)(uintptr_t)p;
}

// copy 64 contiguous bytes from global to LDS for this lane.
// inst_offset is added to BOTH the global and LDS addresses (ISA §10.7/15.18).
__device__ __forceinline__ void async_copy64(unsigned lds, const void* g) {
  asm volatile(
      "global_load_async_to_lds_b128 %0, %1, off\n\t"
      "global_load_async_to_lds_b128 %0, %1, off offset:16\n\t"
      "global_load_async_to_lds_b128 %0, %1, off offset:32\n\t"
      "global_load_async_to_lds_b128 %0, %1, off offset:48"
      :: "v"(lds), "v"(g) : "memory");
}

__device__ __forceinline__ void wait_async0() {
#if __has_builtin(__builtin_amdgcn_s_wait_asynccnt)
  __builtin_amdgcn_s_wait_asynccnt(0);
#else
  asm volatile("s_wait_asynccnt 0x0" ::: "memory");
#endif
}

// ---------------------------------------------------------------------------
// fp32 -> f16 cast, 4 elements per thread
// ---------------------------------------------------------------------------
__global__ __launch_bounds__(256) void cast4_kernel(const float* __restrict__ src,
                                                    half_t* __restrict__ dst, int n4) {
  int i = blockIdx.x * blockDim.x + threadIdx.x;
  if (i < n4) {
    float4 f = ((const float4*)src)[i];
    v4h h;
    h.x = (half_t)f.x; h.y = (half_t)f.y; h.z = (half_t)f.z; h.w = (half_t)f.w;
    ((v4h*)dst)[i] = h;
  }
}

// ---------------------------------------------------------------------------
// GEMM: out[m,n] = sum_k A[m,k] * W[n,k] + bias[n]
// A: [M,1024] f16 row-major, W: [1024,1024] f16 row-major (torch [out,in])
// scatter==1: out is f16 [B,H,T,D]   (for Q/K/V)
// scatter==0: out is f32 [M,1024]    (final projection)
// Block: 256 thr (8 waves). Tile 128x128, K-step 64, double-buffered LDS,
// tiles staged with global_load_async_to_lds_b128.
// ---------------------------------------------------------------------------
#define GK 64
#define GP (GK + 8)           // padded LDS row pitch (144B -> conflict-free)
__global__ __launch_bounds__(256)
void gemm_f16_kernel(const half_t* __restrict__ A, const half_t* __restrict__ W,
                     const float* __restrict__ bias, void* __restrict__ outp,
                     int scatter) {
  __shared__ half_t sA[2][128][GP];
  __shared__ half_t sB[2][128][GP];

  const int tid  = threadIdx.x;
  const int wid  = tid >> 5;
  const int lane = tid & 31;
  const int m0   = blockIdx.x * 128;
  const int n0   = blockIdx.y * 128;
  const int wm   = (wid & 3) * 32;   // wave M offset inside tile
  const int wn   = (wid >> 2) * 64;  // wave N offset inside tile

  const v8f zf = {};
  v8f acc[2][4];
  for (int i = 0; i < 2; i++)
    for (int j = 0; j < 4; j++) acc[i][j] = zf;

  // cooperative staging: thread copies 64B of A and 64B of B per k-tile
  const int lr = tid >> 1;
  const int lc = (tid & 1) * 32;
  const half_t* gA = A + (size_t)(m0 + lr) * HIDDEN + lc;
  const half_t* gB = W + (size_t)(n0 + lr) * HIDDEN + lc;
  const unsigned la0 = lds_addr_of(&sA[0][lr][lc]);
  const unsigned la1 = lds_addr_of(&sA[1][lr][lc]);
  const unsigned lb0 = lds_addr_of(&sB[0][lr][lc]);
  const unsigned lb1 = lds_addr_of(&sB[1][lr][lc]);

  // prologue: async-stage k-tile 0 into buffer 0
  async_copy64(la0, gA);
  async_copy64(lb0, gB);
  wait_async0();

  constexpr int NT = HIDDEN / GK;   // 16 k-tiles
  for (int kt = 0; kt < NT; ++kt) {
    const int cur = kt & 1;
    __syncthreads();   // waits dscnt: all reads of the other buffer are done

    // launch async staging of tile kt+1; overlaps the WMMA block below
    if (kt + 1 < NT) {
      async_copy64(cur ? la0 : la1, gA + (size_t)(kt + 1) * GK);
      async_copy64(cur ? lb0 : lb1, gB + (size_t)(kt + 1) * GK);
    }

    // compute on buffer `cur`: 2 sub-steps of K=32 -> 16 WMMAs
    for (int ks = 0; ks < 2; ++ks) {
      v16h bf[4];
      for (int j = 0; j < 4; j++) {
        int r = wn + j * 16 + (lane & 15);
        int c = ks * 32 + (lane >> 4) * 8;
        v16h b;
        ((uint4*)&b)[0] = *(const uint4*)(&sB[cur][r][c]);
        ((uint4*)&b)[1] = *(const uint4*)(&sB[cur][r][c + 16]);
        bf[j] = b;
      }
      for (int i = 0; i < 2; i++) {
        int r   = wm + i * 16 + (lane & 15);
        int off = ks * 32 + (lane >> 4) * 8;
        v16h a;
        ((uint4*)&a)[0] = *(const uint4*)(&sA[cur][r][off]);
        ((uint4*)&a)[1] = *(const uint4*)(&sA[cur][r][off + 16]);
        for (int j = 0; j < 4; j++)
          acc[i][j] = __builtin_amdgcn_wmma_f32_16x16x32_f16(
              false, a, false, bf[j], (short)0, acc[i][j], false, false);
      }
    }

    // tile kt+1 must be resident before the next barrier publishes it
    if (kt + 1 < NT) wait_async0();
  }

  // epilogue: C/D layout -> (M = e + (lane>>4)*8, N = lane&15)
  const int hi = lane >> 4;
  const int ln = lane & 15;
  for (int i = 0; i < 2; i++) {
    for (int j = 0; j < 4; j++) {
      int ng = n0 + wn + j * 16 + ln;
      float bv = bias[ng];
      for (int e = 0; e < 8; e++) {
        int   mg = m0 + wm + i * 16 + e + hi * 8;
        float v  = acc[i][j][e] + bv;
        if (scatter) {
          int b = mg >> 11, t = mg & (SEQ - 1);
          int h = ng >> 6,  d = ng & (HEAD_DIM - 1);
          ((half_t*)outp)[(((size_t)(b * HEADS + h) * SEQ + t) << 6) + d] = (half_t)v;
        } else {
          ((float*)outp)[(size_t)mg * HIDDEN + ng] = v;
        }
      }
    }
  }
}

// ---------------------------------------------------------------------------
// Flash attention. Grid: (SEQ/64, B*H). Block: 128 thr = 4 waves.
// 64-key tiles in double-buffered LDS; K staged via async copy, V staged
// through VGPRs (needs transpose on the way into LDS).
// ---------------------------------------------------------------------------
#define KT 64
__global__ __launch_bounds__(128)
void attn_kernel(const half_t* __restrict__ Q, const half_t* __restrict__ K,
                 const half_t* __restrict__ V, half_t* __restrict__ Y) {
  __shared__ half_t sK[2][KT][HEAD_DIM + 8];    // keys row-major
  __shared__ half_t sVt[2][HEAD_DIM][KT + 8];   // V transposed [d][key]
  __shared__ half_t sP[4][16][KT + 8];          // per-wave P scratch

  const int tid  = threadIdx.x;
  const int wid  = tid >> 5;
  const int lane = tid & 31;
  const int bh   = blockIdx.y;
  const int b    = bh >> 4;
  const int h    = bh & (HEADS - 1);
  const int q0   = blockIdx.x * 64 + wid * 16;  // this wave's global query base

  const size_t headBase = (size_t)bh * SEQ * HEAD_DIM;

  // staging indices
  const int krow = tid >> 1,  kcol = (tid & 1) * 32;   // K: 64B/thread
  const int vkey = tid & 63,  vdb  = (tid >> 6) * 32;  // V: one key, 32 d-values
  const half_t* gK = K + headBase + (size_t)krow * HEAD_DIM + kcol;
  const half_t* gV = V + headBase + (size_t)vkey * HEAD_DIM + vdb;
  const unsigned lk0 = lds_addr_of(&sK[0][krow][kcol]);
  const unsigned lk1 = lds_addr_of(&sK[1][krow][kcol]);

  // Q fragments (A operand), 2 k-steps over D=64, kept in registers
  v16h qf[2];
  {
    const half_t* qrow = Q + headBase + (size_t)(q0 + (lane & 15)) * HEAD_DIM;
    int off = (lane >> 4) * 8;
    for (int s = 0; s < 2; s++) {
      v16h a;
      ((uint4*)&a)[0] = *(const uint4*)(qrow + s * 32 + off);
      ((uint4*)&a)[1] = *(const uint4*)(qrow + s * 32 + 16 + off);
      qf[s] = a;
    }
  }

  const v8f zf = {};
  v8f o[4];
  for (int j = 0; j < 4; j++) o[j] = zf;
  float m8[8], l8[8];
  for (int e = 0; e < 8; e++) { m8[e] = -1e30f; l8[e] = 0.f; }

  const float scale = 0.125f;  // 1/sqrt(64)

  // prologue: stage key-tile 0 into buffer 0
  async_copy64(lk0, gK);
  {
    const uint4* gv = (const uint4*)gV;
    uint4 tmp[4];
    tmp[0] = gv[0]; tmp[1] = gv[1]; tmp[2] = gv[2]; tmp[3] = gv[3];
    const half_t* hp = (const half_t*)tmp;
    for (int d = 0; d < 32; d++) sVt[0][vdb + d][vkey] = hp[d];
  }
  wait_async0();

  constexpr int NKT = SEQ / KT;   // 32 key tiles
  for (int kt = 0; kt < NKT; kt++) {
    const int cur = kt & 1, nxt = cur ^ 1;
    __syncthreads();

    // stage tile kt+1: K async into LDS, V into regs (transposed store later)
    uint4 rv[4];
    if (kt + 1 < NKT) {
      async_copy64(cur ? lk0 : lk1, gK + (size_t)(kt + 1) * KT * HEAD_DIM);
      const uint4* gv = (const uint4*)(gV + (size_t)(kt + 1) * KT * HEAD_DIM);
      rv[0] = gv[0]; rv[1] = gv[1]; rv[2] = gv[2]; rv[3] = gv[3];
    }

    // S = Q @ K^T  (B operand: lane n = key row, contiguous over d)
    v8f sacc[4];
    for (int j = 0; j < 4; j++) sacc[j] = zf;
    for (int ds = 0; ds < 2; ds++) {
      for (int j = 0; j < 4; j++) {
        int r = j * 16 + (lane & 15);
        int c = ds * 32 + (lane >> 4) * 8;
        v16h kb;
        ((uint4*)&kb)[0] = *(const uint4*)(&sK[cur][r][c]);
        ((uint4*)&kb)[1] = *(const uint4*)(&sK[cur][r][c + 16]);
        sacc[j] = __builtin_amdgcn_wmma_f32_16x16x32_f16(
            false, qf[ds], false, kb, (short)0, sacc[j], false, false);
      }
    }

    // online softmax; row m = e + (lane>>4)*8 lives in 16 lanes of one half
    for (int e = 0; e < 8; e++) {
      float mx = -1e30f;
      for (int j = 0; j < 4; j++) mx = fmaxf(mx, sacc[j][e]);
      for (int msk = 8; msk >= 1; msk >>= 1)
        mx = fmaxf(mx, __shfl_xor(mx, msk, 32));
      mx *= scale;
      float nm    = fmaxf(m8[e], mx);
      float alpha = expf(m8[e] - nm);
      int   prow  = e + (lane >> 4) * 8;
      float rsum  = 0.f;
      for (int j = 0; j < 4; j++) {
        float p = expf(sacc[j][e] * scale - nm);
        rsum += p;
        sP[wid][prow][j * 16 + (lane & 15)] = (half_t)p;
      }
      for (int msk = 8; msk >= 1; msk >>= 1)
        rsum += __shfl_xor(rsum, msk, 32);
      l8[e] = l8[e] * alpha + rsum;
      m8[e] = nm;
      for (int j = 0; j < 4; j++) o[j][e] = o[j][e] * alpha;
    }
    // no barrier needed: sP is per-wave and same-wave LDS ops are in-order

    // O += P @ V   (B operand from transposed V: contiguous over keys)
    for (int ks = 0; ks < 2; ks++) {
      int r   = lane & 15;
      int off = ks * 32 + (lane >> 4) * 8;
      v16h pa;
      ((uint4*)&pa)[0] = *(const uint4*)(&sP[wid][r][off]);
      ((uint4*)&pa)[1] = *(const uint4*)(&sP[wid][r][off + 16]);
      for (int j = 0; j < 4; j++) {
        int vr = j * 16 + (lane & 15);       // d
        int vc = ks * 32 + (lane >> 4) * 8;  // key
        v16h vb;
        ((uint4*)&vb)[0] = *(const uint4*)(&sVt[cur][vr][vc]);
        ((uint4*)&vb)[1] = *(const uint4*)(&sVt[cur][vr][vc + 16]);
        o[j] = __builtin_amdgcn_wmma_f32_16x16x32_f16(
            false, pa, false, vb, (short)0, o[j], false, false);
      }
    }

    // finish staging tile kt+1 (transpose V, drain async K copy)
    if (kt + 1 < NKT) {
      const half_t* hp = (const half_t*)rv;
      for (int d = 0; d < 32; d++) sVt[nxt][vdb + d][vkey] = hp[d];
      wait_async0();
    }
  }

  // normalize and emit y as f16 [B, T, H*D] (row-major A for the final GEMM)
  const int hi = lane >> 4, ln = lane & 15;
  for (int j = 0; j < 4; j++) {
    for (int e = 0; e < 8; e++) {
      int   t   = q0 + e + hi * 8;
      int   d   = j * 16 + ln;
      float val = o[j][e] / l8[e];
      Y[((size_t)(b * SEQ + t)) * HIDDEN + h * HEAD_DIM + d] = (half_t)val;
    }
  }
}

// ---------------------------------------------------------------------------
extern "C" void kernel_launch(void* const* d_in, const int* in_sizes, int n_in,
                              void* d_out, int out_size, void* d_ws, size_t ws_size,
                              hipStream_t stream) {
  const float* x  = (const float*)d_in[0];
  const float* Wq = (const float*)d_in[1];
  const float* bq = (const float*)d_in[2];
  const float* Wk = (const float*)d_in[3];
  const float* bk = (const float*)d_in[4];
  const float* Wv = (const float*)d_in[5];
  const float* bv = (const float*)d_in[6];
  const float* Wo = (const float*)d_in[7];
  const float* bo = (const float*)d_in[8];
  float* out = (float*)d_out;

  half_t* xh  = (half_t*)d_ws;
  half_t* Wqh = xh  + (size_t)MTOT * HIDDEN;
  half_t* Wkh = Wqh + (size_t)HIDDEN * HIDDEN;
  half_t* Wvh = Wkh + (size_t)HIDDEN * HIDDEN;
  half_t* Woh = Wvh + (size_t)HIDDEN * HIDDEN;
  half_t* Qh  = Woh + (size_t)HIDDEN * HIDDEN;
  half_t* Kh  = Qh  + (size_t)MTOT * HIDDEN;
  half_t* Vh  = Kh  + (size_t)MTOT * HIDDEN;
  half_t* Yh  = Vh  + (size_t)MTOT * HIDDEN;

  const int nx4 = MTOT * HIDDEN / 4;
  const int nw4 = HIDDEN * HIDDEN / 4;
  cast4_kernel<<<(nx4 + 255) / 256, 256, 0, stream>>>(x,  xh,  nx4);
  cast4_kernel<<<(nw4 + 255) / 256, 256, 0, stream>>>(Wq, Wqh, nw4);
  cast4_kernel<<<(nw4 + 255) / 256, 256, 0, stream>>>(Wk, Wkh, nw4);
  cast4_kernel<<<(nw4 + 255) / 256, 256, 0, stream>>>(Wv, Wvh, nw4);
  cast4_kernel<<<(nw4 + 255) / 256, 256, 0, stream>>>(Wo, Woh, nw4);

  dim3 gg(MTOT / 128, HIDDEN / 128);
  gemm_f16_kernel<<<gg, 256, 0, stream>>>(xh, Wqh, bq, (void*)Qh, 1);
  gemm_f16_kernel<<<gg, 256, 0, stream>>>(xh, Wkh, bk, (void*)Kh, 1);
  gemm_f16_kernel<<<gg, 256, 0, stream>>>(xh, Wvh, bv, (void*)Vh, 1);

  dim3 ga(SEQ / 64, BATCH * HEADS);
  attn_kernel<<<ga, 128, 0, stream>>>(Qh, Kh, Vh, Yh);

  gemm_f16_kernel<<<gg, 256, 0, stream>>>(Yh, Woh, bo, (void*)out, 0);
}
